// MaskedMultiHeadSelfAttention_21414706938579
// MI455X (gfx1250) — matrix operand order repaired and checked by reading, not compiled, not run on hardware
//
#include <hip/hip_runtime.h>

typedef __bf16 bf16;
typedef __attribute__((ext_vector_type(16))) __bf16 v16bf;
typedef __attribute__((ext_vector_type(8)))  __bf16 v8bf;
typedef __attribute__((ext_vector_type(8)))  float  v8f;
typedef __attribute__((ext_vector_type(4)))  float  f32x4;

#define B_  2
#define C_  2048
#define E_  1024
#define H_  16
#define HD_ 64
#define MTOT (B_*C_)      // 4096 rows
#define NQKV (3*E_)       // 3072

// ---------------------------------------------------------------------------
// WMMA fragment layouts (wave32, CDNA5 ISA 7.12.2):
//  A 16x32 bf16 : lane L -> row M = L&15 ; element i -> K = (i<8?i:i+8) + (L>=16?8:0)
//  B 32x16 bf16 : lane L -> col N = L&15 ; K order as A; this code uses K = i + (L>=16?16:0) for both A and B
//  C/D 16x16 f32: VGPR r, lane L -> M = r + 8*(L>=16), N = L&15
// ---------------------------------------------------------------------------

__device__ __forceinline__ v16bf cvt16(f32x4 r0, f32x4 r1, f32x4 r2, f32x4 r3) {
    v16bf o;
#pragma unroll
    for (int i = 0; i < 4; ++i) {
        o[i]      = (bf16)r0[i];
        o[4 + i]  = (bf16)r1[i];
        o[8 + i]  = (bf16)r2[i];
        o[12 + i] = (bf16)r3[i];
    }
    return o;
}
__device__ __forceinline__ v16bf cat8(v8bf a, v8bf b) {
    return __builtin_shufflevector(a, b, 0,1,2,3,4,5,6,7,8,9,10,11,12,13,14,15);
}

// GEMM: qkv = x @ w_qkv.T + b_qkv ; scatter to Q,K [B,H,C,HD] and V^T [B,H,HD,C]
__global__ __launch_bounds__(128) void qkv_proj_kernel(
    const float* __restrict__ x, const float* __restrict__ w,
    const float* __restrict__ bias,
    bf16* __restrict__ qb, bf16* __restrict__ kb, bf16* __restrict__ vb)
{
    const int lane = threadIdx.x & 31;
    const int wave = threadIdx.x >> 5;
    const int gwave = blockIdx.x * 4 + wave;
    const int n64t = NQKV / 64;              // 48
    const int mtile = gwave / n64t;
    const int n64   = gwave % n64t;
    if (mtile >= MTOT / 16) return;          // wave-uniform

    const int hi  = lane >> 4;
    const int l15 = lane & 15;
    const int arow = mtile * 16 + l15;

    // base pointers (f32x4 units); per k-step (32 floats) advance by 8
    const f32x4* xp = (const f32x4*)(x + (size_t)arow * E_) + hi * 4;
    const f32x4* wp[4];
#pragma unroll
    for (int t = 0; t < 4; ++t)
        wp[t] = (const f32x4*)(w + (size_t)(n64 * 64 + t * 16 + l15) * E_) + hi * 4;

    f32x4 ar[2][4];
    f32x4 br[2][4][4];
    v8f acc[4] = {v8f{0.f},v8f{0.f},v8f{0.f},v8f{0.f}};

    auto LOAD = [&](int pb, int s) {
        const f32x4* xq = xp + s * 8;
        ar[pb][0] = xq[0]; ar[pb][1] = xq[1];    // K 16h+0..7
        ar[pb][2] = xq[2]; ar[pb][3] = xq[3];    // K 16h+8..15
#pragma unroll
        for (int t = 0; t < 4; ++t) {
            const f32x4* wq = wp[t] + s * 8;
            br[pb][t][0] = wq[0]; br[pb][t][1] = wq[1];
            br[pb][t][2] = wq[2]; br[pb][t][3] = wq[3];
        }
    };
    auto COMP = [&](int pb) {
        v16bf a = cvt16(ar[pb][0], ar[pb][1], ar[pb][2], ar[pb][3]);
#pragma unroll
        for (int t = 0; t < 4; ++t) {
            v16bf bfr = cvt16(br[pb][t][0], br[pb][t][1], br[pb][t][2], br[pb][t][3]);
            acc[t] = __builtin_amdgcn_wmma_f32_16x16x32_bf16(
                false, a, false, bfr, (short)0, acc[t], false, false);
        }
    };

    LOAD(0, 0);
    for (int s = 0; s < E_ / 32; s += 2) {   // 32 steps, double-buffered
        LOAD(1, s + 1);
        COMP(0);
        if (s + 2 < E_ / 32) LOAD(0, s + 2);
        COMP(1);
    }

    // epilogue: bias + scatter to q/k/v
#pragma unroll
    for (int t = 0; t < 4; ++t) {
        const int n = n64 * 64 + t * 16 + l15;
        const int which = n >> 10;           // uniform per 16-col tile
        const int e = n & 1023;
        const int h = e >> 6, d = e & 63;
        const float bv = bias[n];
#pragma unroll
        for (int r = 0; r < 8; ++r) {
            const int m  = mtile * 16 + r + 8 * hi;
            const int bb = m >> 11, c = m & (C_ - 1);
            const float val = acc[t][r] + bv;
            if (which == 0)
                qb[(((size_t)bb * H_ + h) * C_ + c) * HD_ + d] = (bf16)val;
            else if (which == 1)
                kb[(((size_t)bb * H_ + h) * C_ + c) * HD_ + d] = (bf16)val;
            else
                vb[(((size_t)bb * H_ + h) * HD_ + d) * C_ + c] = (bf16)val;
        }
    }
}

// Flash attention: one wave per (b,h,qtile16); key blocks of 32.
__global__ __launch_bounds__(128) void attn_kernel(
    const bf16* __restrict__ q, const bf16* __restrict__ k,
    const bf16* __restrict__ v, const unsigned char* __restrict__ pad,
    bf16* __restrict__ ao)
{
    __shared__ __align__(16) bf16 plds[4][16][32];
    const int lane = threadIdx.x & 31;
    const int wave = threadIdx.x >> 5;
    const int gwave = blockIdx.x * 4 + wave;
    const int qt = gwave & 127;              // C_/16 = 128
    const int bh = gwave >> 7;
    if (bh >= B_ * H_) return;
    const int h = bh & (H_ - 1), b = bh >> 4;

    const int hi = lane >> 4;
    const int l15 = lane & 15;
    const size_t base_ck = ((size_t)b * H_ + h) * C_;   // q,k: [.,C,HD]
    const size_t base_dc = ((size_t)b * H_ + h) * HD_;  // vT:  [.,HD,C]

    // Q fragments (A layout); HD=64 -> two 32-wide K chunks
    v16bf qa[2];
    const int qrow = qt * 16 + l15;
    {
        const v8bf* qp = (const v8bf*)(q + (base_ck + qrow) * HD_) + hi * 2;
        qa[0] = cat8(qp[0], qp[1]);          // K 0..31  (16 contiguous elems at 16h)
        qa[1] = cat8(qp[4], qp[5]);          // K 32..63
    }

    float m_i[8], l_i[8];
#pragma unroll
    for (int r = 0; r < 8; ++r) { m_i[r] = -1e30f; l_i[r] = 0.f; }
    v8f oacc[4] = {v8f{0.f},v8f{0.f},v8f{0.f},v8f{0.f}};

    const float scale = 0.125f;              // 1/sqrt(64)
    const int kend = qt * 16 + 16;           // causal bound

    // per-lane V row pointers (d fixed per lane per t2)
    const v8bf* vrow[4];
#pragma unroll
    for (int t2 = 0; t2 < 4; ++t2)
        vrow[t2] = (const v8bf*)(v + (base_dc + t2 * 16 + l15) * C_) + hi * 2;

    for (int kbase = 0; kbase < kend; kbase += 32) {
        // ---- issue ALL raw loads for this block up front ----
        v8bf kraw[2][2][2];                  // [tile][kchunk][half]
#pragma unroll
        for (int t = 0; t < 2; ++t) {
            const v8bf* kp = (const v8bf*)(k + (base_ck + kbase + t * 16 + l15) * HD_) + hi * 2;
#pragma unroll
            for (int kc2 = 0; kc2 < 2; ++kc2) {
                kraw[t][kc2][0] = kp[kc2 * 4 + 0];
                kraw[t][kc2][1] = kp[kc2 * 4 + 1];
            }
        }
        v8bf vraw[4][2];                     // consumed only after softmax
#pragma unroll
        for (int t2 = 0; t2 < 4; ++t2) {
            const v8bf* vp = vrow[t2] + (kbase >> 3); // kbase elems -> /8 v8bf
            vraw[t2][0] = vp[0];
            vraw[t2][1] = vp[1];
        }
        // prefetch next key block (global_prefetch_b8)
        __builtin_prefetch((const void*)(k + (base_ck + kbase + 32 + l15) * HD_), 0, 1);

        // ---- scores: two 16x16 tiles ----
        v8f s[2] = {v8f{0.f}, v8f{0.f}};
#pragma unroll
        for (int t = 0; t < 2; ++t)
#pragma unroll
            for (int kc2 = 0; kc2 < 2; ++kc2)
                s[t] = __builtin_amdgcn_wmma_f32_16x16x32_bf16(
                    false, qa[kc2], false, cat8(kraw[t][kc2][0], kraw[t][kc2][1]),
                    (short)0, s[t], false, false);

        // ---- mask + online softmax (row r lives in VGPR r across the 16-lane group) ----
        const int key0 = kbase + l15, key1 = key0 + 16;
        const bool p0ok = pad[b * C_ + key0] != 0;
        const bool p1ok = pad[b * C_ + key1] != 0;
#pragma unroll
        for (int r = 0; r < 8; ++r) {
            const int qg = qt * 16 + r + 8 * hi;
            float a0 = (p0ok && key0 <= qg) ? s[0][r] * scale : -1e9f;
            float a1 = (p1ok && key1 <= qg) ? s[1][r] * scale : -1e9f;
            float mx = fmaxf(a0, a1);
#pragma unroll
            for (int o = 1; o < 16; o <<= 1) mx = fmaxf(mx, __shfl_xor(mx, o, 32));
            const float mnew  = fmaxf(m_i[r], mx);
            const float alpha = __expf(m_i[r] - mnew);
            m_i[r] = mnew;
            const float p0 = __expf(a0 - mnew);
            const float p1 = __expf(a1 - mnew);
            float ps = p0 + p1;
#pragma unroll
            for (int o = 1; o < 16; o <<= 1) ps += __shfl_xor(ps, o, 32);
            l_i[r] = l_i[r] * alpha + ps;
#pragma unroll
            for (int t2 = 0; t2 < 4; ++t2) oacc[t2][r] *= alpha;
            plds[wave][r + 8 * hi][l15]      = (bf16)p0;   // D-layout -> LDS
            plds[wave][r + 8 * hi][16 + l15] = (bf16)p1;
        }
        // ---- P back as A fragment via two ds_load_b128 ----
        const v8bf* pp = (const v8bf*)(&plds[wave][l15][0]) + hi * 2;
        v16bf pa = cat8(pp[0], pp[1]);

        // ---- PV (V^T rows contiguous in key) ----
#pragma unroll
        for (int t2 = 0; t2 < 4; ++t2)
            oacc[t2] = __builtin_amdgcn_wmma_f32_16x16x32_bf16(
                false, pa, false, cat8(vraw[t2][0], vraw[t2][1]),
                (short)0, oacc[t2], false, false);
    }
    // ---- finalize: out[b][c][h*64+d] ----
#pragma unroll
    for (int t2 = 0; t2 < 4; ++t2) {
        const int d = t2 * 16 + l15;
#pragma unroll
        for (int r = 0; r < 8; ++r) {
            const int m = qt * 16 + r + 8 * hi;
            ao[((size_t)b * C_ + m) * E_ + h * HD_ + d] = (bf16)(oacc[t2][r] / l_i[r]);
        }
    }
}

// GEMM: out = attn @ w_out.T + b_out  (f32 output)
__global__ __launch_bounds__(128) void out_proj_kernel(
    const bf16* __restrict__ a_in, const float* __restrict__ w,
    const float* __restrict__ bias, float* __restrict__ out)
{
    const int lane = threadIdx.x & 31;
    const int wave = threadIdx.x >> 5;
    const int gwave = blockIdx.x * 4 + wave;
    const int n64t = E_ / 64;                // 16
    const int mtile = gwave / n64t;
    const int n64   = gwave % n64t;
    if (mtile >= MTOT / 16) return;

    const int hi = lane >> 4;
    const int l15 = lane & 15;
    const int arow = mtile * 16 + l15;

    const v8bf* ap = (const v8bf*)(a_in + (size_t)arow * E_) + hi * 2;
    const f32x4* wp[4];
#pragma unroll
    for (int t = 0; t < 4; ++t)
        wp[t] = (const f32x4*)(w + (size_t)(n64 * 64 + t * 16 + l15) * E_) + hi * 4;

    v8bf ar[2][2];
    f32x4 br[2][4][4];
    v8f acc[4] = {v8f{0.f},v8f{0.f},v8f{0.f},v8f{0.f}};

    auto LOAD = [&](int pb, int s) {
        const v8bf* aq = ap + s * 4;
        ar[pb][0] = aq[0];                   // K 16h+0..7
        ar[pb][1] = aq[1];                   // K 16h+8..15
#pragma unroll
        for (int t = 0; t < 4; ++t) {
            const f32x4* wq = wp[t] + s * 8;
            br[pb][t][0] = wq[0]; br[pb][t][1] = wq[1];
            br[pb][t][2] = wq[2]; br[pb][t][3] = wq[3];
        }
    };
    auto COMP = [&](int pb) {
        v16bf a = cat8(ar[pb][0], ar[pb][1]);
#pragma unroll
        for (int t = 0; t < 4; ++t) {
            v16bf bfr = cvt16(br[pb][t][0], br[pb][t][1], br[pb][t][2], br[pb][t][3]);
            acc[t] = __builtin_amdgcn_wmma_f32_16x16x32_bf16(
                false, a, false, bfr, (short)0, acc[t], false, false);
        }
    };

    LOAD(0, 0);
    for (int s = 0; s < E_ / 32; s += 2) {
        LOAD(1, s + 1);
        COMP(0);
        if (s + 2 < E_ / 32) LOAD(0, s + 2);
        COMP(1);
    }

#pragma unroll
    for (int t = 0; t < 4; ++t) {
        const int n = n64 * 64 + t * 16 + l15;
        const float bv = bias[n];
#pragma unroll
        for (int r = 0; r < 8; ++r) {
            const int m = mtile * 16 + r + 8 * hi;
            out[(size_t)m * E_ + n] = acc[t][r] + bv;
        }
    }
}

extern "C" void kernel_launch(void* const* d_in, const int* in_sizes, int n_in,
                              void* d_out, int out_size, void* d_ws, size_t ws_size,
                              hipStream_t stream) {
    (void)in_sizes; (void)n_in; (void)out_size; (void)ws_size;
    const float*         x     = (const float*)d_in[0];
    const unsigned char* pad   = (const unsigned char*)d_in[1];
    const float*         w_qkv = (const float*)d_in[2];
    const float*         b_qkv = (const float*)d_in[3];
    const float*         w_out = (const float*)d_in[4];
    const float*         b_out = (const float*)d_in[5];
    float*               out   = (float*)d_out;

    const size_t nqk = (size_t)B_ * H_ * C_ * HD_;   // 4,194,304 elems
    bf16* qbuf = (bf16*)d_ws;
    bf16* kbuf = qbuf + nqk;
    bf16* vbuf = kbuf + nqk;                          // stored [B,H,HD,C]
    bf16* abuf = vbuf + nqk;                          // attn out [B,C,E]

    qkv_proj_kernel<<<3072, 128, 0, stream>>>(x, w_qkv, b_qkv, qbuf, kbuf, vbuf);
    attn_kernel<<<1024, 128, 0, stream>>>(qbuf, kbuf, vbuf, pad, abuf);
    out_proj_kernel<<<1024, 128, 0, stream>>>(abuf, w_out, b_out, out);
}
